// Seq2SeqLSTM_27676769255798
// MI455X (gfx1250) — compile-verified, matrix operand order checked by reading, not run on hardware
//
#include <hip/hip_runtime.h>
#include <hip/hip_bf16.h>

// ---------------------------------------------------------------------------
// Seq2Seq 2-layer LSTM (B=1024, T=50, C=66, H=1024) for gfx1250 (CDNA5).
// bf16 WMMA (v_wmma_f32_16x16x32_bf16) + f32 accumulation.  Fragment-major
// pre-scatter so every lane loads its 32-byte WMMA slice contiguously.
// Weight slabs streamed to LDS with gfx1250 async global->LDS (ASYNCcnt),
// ring of 4 slots, prefetch distance 2, one barrier per K-slice.
// ---------------------------------------------------------------------------

#define B_  1024
#define T_  50
#define C_  66
#define H_  1024
#define KXP 96            // C padded to multiple of 32
#define N4  (4 * H_)      // 4096 gate columns
#define FCNP 80           // fc output cols padded to multiple of 16

typedef __attribute__((ext_vector_type(16))) __bf16 v16bf;
typedef __attribute__((ext_vector_type(8)))  float  v8f;

#if defined(__HIP_DEVICE_COMPILE__) && __has_builtin(__builtin_amdgcn_global_load_async_to_lds_b128)
#define USE_ASYNC_LDS 1
#else
#define USE_ASYNC_LDS 0
#endif

#if USE_ASYNC_LDS
typedef int v4i_ls __attribute__((vector_size(16)));
typedef __attribute__((address_space(1))) v4i_ls* as1_v4i;
typedef __attribute__((address_space(3))) v4i_ls* as3_v4i;
#if __has_builtin(__builtin_amdgcn_s_wait_asynccnt)
#define ASYNC_WAIT(n) __builtin_amdgcn_s_wait_asynccnt(n)
#else
#define ASYNC_WAIT(n) asm volatile("s_wait_asynccnt %0" ::"n"(n) : "memory")
#endif
#endif

// Element position inside a 512-element fragment for matrix coord (mm, kk),
// mm in [0,16), kk in [0,32): CDNA5 16-bit A/B fragment register order.
__device__ __forceinline__ int frag_pos(int mm, int kk) {
  int lane = mm + (((kk & 15) >= 8) ? 16 : 0);
  int v    = ((kk >= 16) ? 4 : 0) + ((kk & 7) >> 1);
  return lane * 16 + v * 2 + (kk & 1);
}

__device__ __forceinline__ float sigm(float x) {
  return 1.0f / (1.0f + __expf(-x));
}
__device__ __forceinline__ float tanh_fast(float x) {
  float ax = fabsf(x);
  float e  = __expf(-2.0f * ax);
  float t  = (1.0f - e) / (1.0f + e);
  return copysignf(t, x);
}

#define WMMA_BF16(a, b, c) \
  __builtin_amdgcn_wmma_f32_16x16x32_bf16(false, (a), false, (b), (short)0, (c), false, false)

// ------------------------------ prologue -----------------------------------

__global__ void zero_u32(unsigned* __restrict__ p, long n) {
  long i = (long)blockIdx.x * blockDim.x + threadIdx.x;
  long s = (long)gridDim.x * blockDim.x;
  for (; i < n; i += s) p[i] = 0u;
}

__global__ void scatter_w(const float* __restrict__ W, int N, int Npad,
                          int K, int Kpad, __bf16* __restrict__ out) {
  int idx = blockIdx.x * blockDim.x + threadIdx.x;
  int total = Npad * Kpad;
  if (idx >= total) return;
  int n = idx / Kpad, k = idx % Kpad;
  float v = (n < N && k < K) ? W[(size_t)n * K + k] : 0.0f;
  int KF = Kpad >> 5;
  size_t o = (((size_t)(n >> 4) * KF + (k >> 5)) << 9) + frag_pos(n & 15, k & 31);
  out[o] = (__bf16)v;
}

__global__ void scatter_x(const float* __restrict__ x, __bf16* __restrict__ out) {
  long idx = (long)blockIdx.x * blockDim.x + threadIdx.x;
  long total = (long)B_ * T_ * KXP;
  if (idx >= total) return;
  int k = (int)(idx % KXP);
  int b = (int)((idx / KXP) % B_);
  int t = (int)(idx / ((long)KXP * B_));
  float v = (k < C_) ? x[((size_t)b * T_ + t) * C_ + k] : 0.0f;
  size_t o = (size_t)t * ((B_ / 16) * (KXP / 32) * 512) +
             (((size_t)(b >> 4) * (KXP / 32) + (k >> 5)) << 9) +
             frag_pos(b & 15, k & 31);
  out[o] = (__bf16)v;
}

__global__ void init_dec(const float* __restrict__ x, float* __restrict__ dec) {
  int i = blockIdx.x * blockDim.x + threadIdx.x;
  if (i >= B_ * C_) return;
  int b = i / C_, j = i % C_;
  dec[i] = x[((size_t)b * T_ + (T_ - 1)) * C_ + j];
}

// --------------------------- fused LSTM layer ------------------------------
// g = Ax@Wih^T + Ah@Whh^T + bias ; gate math ; c update in place ; h_new
// re-scattered to bf16 A-fragments.  Grid (B/128)*(H/16) = 512 blocks,
// 128 threads (4 waves).  Wave owns TWO 16-row M-tiles -> 8 WMMA per K-slice
// against the 4 gate B-fragments staged in LDS (shared by all waves).
// KFX (x-input K-slices: 3 for layer0, 32 for layer1) is a template constant
// so all trip counts / offsets strength-reduce.
template <int KFX>
__global__ __launch_bounds__(128) void lstm_layer(
    const __bf16* __restrict__ Axf,
    const __bf16* __restrict__ Ahf,
    const __bf16* __restrict__ Bih,
    const __bf16* __restrict__ Bhh,
    const float* __restrict__ bias,
    float* __restrict__ c,
    __bf16* __restrict__ Hout)
{
  constexpr int KFH = H_ / 32;           // 32 h K-slices
  constexpr int KFT = KFX + KFH;
  __shared__ __align__(32) __bf16 lds[4 * 2048];   // ring of 4 slabs (16 KB)

  const int mg   = blockIdx.x & 7;                 // 128-row group
  const int nt   = blockIdx.x >> 3;                // column tile within H
  const int wave = threadIdx.x >> 5;
  const int lane = threadIdx.x & 31;
  const int mt0  = mg * 8 + wave * 2;              // first of 2 row tiles

  v8f acc[2][4];
#pragma unroll
  for (int t = 0; t < 2; ++t)
#pragma unroll
    for (int g = 0; g < 4; ++g) acc[t][g] = (v8f){0, 0, 0, 0, 0, 0, 0, 0};

  const int ntw = nt + wave * (H_ / 16);           // wave w stages gate w
  const __bf16* bX  = Bih + (size_t)ntw * KFX * 512 + (lane << 4);
  const __bf16* bH  = Bhh + (size_t)ntw * KFH * 512 + (lane << 4);
  const __bf16* aX0 = Axf + (size_t)mt0 * KFX * 512 + (lane << 4);
  const __bf16* aH0 = Ahf + (size_t)mt0 * KFH * 512 + (lane << 4);
  const __bf16* aX1 = aX0 + (size_t)KFX * 512;     // mt0+1 panel is contiguous
  const __bf16* aH1 = aH0 + (size_t)KFH * 512;

  auto consume_slot = [&](int slot, const __bf16* a0p, const __bf16* a1p) {
    const __bf16* lb = lds + (size_t)slot * 2048;
    const v16bf a0 = *(const v16bf*)a0p;
    const v16bf a1 = *(const v16bf*)a1p;
    const v16bf b0 = *(const v16bf*)(lb +    0 + (lane << 4));
    const v16bf b1 = *(const v16bf*)(lb +  512 + (lane << 4));
    const v16bf b2 = *(const v16bf*)(lb + 1024 + (lane << 4));
    const v16bf b3 = *(const v16bf*)(lb + 1536 + (lane << 4));
    acc[0][0] = WMMA_BF16(a0, b0, acc[0][0]);
    acc[0][1] = WMMA_BF16(a0, b1, acc[0][1]);
    acc[0][2] = WMMA_BF16(a0, b2, acc[0][2]);
    acc[0][3] = WMMA_BF16(a0, b3, acc[0][3]);
    acc[1][0] = WMMA_BF16(a1, b0, acc[1][0]);
    acc[1][1] = WMMA_BF16(a1, b1, acc[1][1]);
    acc[1][2] = WMMA_BF16(a1, b2, acc[1][2]);
    acc[1][3] = WMMA_BF16(a1, b3, acc[1][3]);
  };

#if USE_ASYNC_LDS
  // --- async global->LDS pipeline: ring 4, prefetch distance 2 -------------
  const __bf16* sp = bX;                 // running stage pointer
  int sidx = 0;                          // next slice index to stage
  auto stage_next = [&](int slot) {
    __bf16* d = lds + (size_t)slot * 2048 + (wave << 9) + (lane << 4);
    __builtin_amdgcn_global_load_async_to_lds_b128((as1_v4i)sp, (as3_v4i)d, 0, 0);
    __builtin_amdgcn_global_load_async_to_lds_b128((as1_v4i)(sp + 8), (as3_v4i)(d + 8), 0, 0);
    ++sidx;
    sp = (sidx == KFX) ? bH : sp + 512;  // uniform phase switch
  };
  stage_next(0);
  stage_next(1);
  const __bf16* a0 = aX0;
  const __bf16* a1 = aX1;
  for (int s = 0; s < KFT - 2; ++s) {
    stage_next((s + 2) & 3);
    ASYNC_WAIT(4);                       // slice s landed; s+1, s+2 in flight
    __syncthreads();
    consume_slot(s & 3, a0, a1);
    a0 = (s + 1 == KFX) ? aH0 : a0 + 512;
    a1 = (s + 1 == KFX) ? aH1 : a1 + 512;
  }
  ASYNC_WAIT(2);
  __syncthreads();
  consume_slot((KFT - 2) & 3, a0, a1);
  ASYNC_WAIT(0);
  __syncthreads();
  consume_slot((KFT - 1) & 3, aH0 + (size_t)(KFH - 1) * 512,
                              aH1 + (size_t)(KFH - 1) * 512);
#else
  // --- sync staging fallback (global -> VGPR -> ds_store) ------------------
  const __bf16* sp = bX;
  const __bf16* a0 = aX0;
  const __bf16* a1 = aX1;
  for (int s = 0; s < KFT; ++s) {
    const v16bf bs = *(const v16bf*)sp;
    __syncthreads();
    *(v16bf*)(lds + (wave << 9) + (lane << 4)) = bs;
    __syncthreads();
    consume_slot(0, a0, a1);
    sp = (s + 1 == KFX) ? bH : sp + 512;
    a0 = (s + 1 == KFX) ? aH0 : a0 + 512;
    a1 = (s + 1 == KFX) ? aH1 : a1 + 512;
  }
#endif

  // epilogue: lane holds col N = lane&15, rows M = v + 8*(lane>=16)
  const int N  = lane & 15;
  const int Mh = (lane >> 4) * 8;
  const int n  = nt * 16 + N;
  const float bi = bias[n];
  const float bf = bias[n + H_];
  const float bg = bias[n + 2 * H_];
  const float bo = bias[n + 3 * H_];
#pragma unroll
  for (int t = 0; t < 2; ++t) {
    const int mt = mt0 + t;
#pragma unroll
    for (int v = 0; v < 8; ++v) {
      int b = mt * 16 + Mh + v;
      float iv = sigm(acc[t][0][v] + bi);
      float fv = sigm(acc[t][1][v] + bf);
      float gv = tanh_fast(acc[t][2][v] + bg);
      float ov = sigm(acc[t][3][v] + bo);
      size_t ci = (size_t)b * H_ + n;
      float cnew = fv * c[ci] + iv * gv;
      c[ci] = cnew;
      float h = ov * tanh_fast(cnew);
      size_t o = (((size_t)(b >> 4) * (H_ / 32) + (n >> 5)) << 9) + frag_pos(b & 15, n & 31);
      Hout[o] = (__bf16)h;
    }
  }
}

// ------------------------------- decoder FC --------------------------------
__global__ __launch_bounds__(128) void fc_gemm(const __bf16* __restrict__ Af,
                                               const __bf16* __restrict__ Bf,
                                               float* __restrict__ delta) {
  const int mg = blockIdx.x, nt = blockIdx.y;
  const int wave = threadIdx.x >> 5, lane = threadIdx.x & 31;
  const int mtile = mg * 4 + wave;
  const __bf16* ap = Af + (size_t)mtile * 32 * 512 + (lane << 4);
  const __bf16* bp = Bf + (size_t)nt    * 32 * 512 + (lane << 4);
  v8f acc = {0, 0, 0, 0, 0, 0, 0, 0};
  for (int kf = 0; kf < H_ / 32; ++kf) {
    v16bf a = *(const v16bf*)(ap + (size_t)kf * 512);
    v16bf b = *(const v16bf*)(bp + (size_t)kf * 512);
    acc = WMMA_BF16(a, b, acc);
  }
  const int N = lane & 15, Mh = (lane >> 4) * 8;
#pragma unroll
  for (int v = 0; v < 8; ++v) {
    int b = mtile * 16 + Mh + v;
    delta[(size_t)b * FCNP + nt * 16 + N] = acc[v];
  }
}

// layernorm(delta + fc_b) + residual ; emit out[:,s,:] ; update dec and
// re-scatter dec_new into A-fragment layout (pad region stays zero).
__global__ void dec_finish(const float* __restrict__ delta, const float* __restrict__ fcb,
                           const float* __restrict__ alpha, const float* __restrict__ beta,
                           float* __restrict__ dec, float* __restrict__ out, int s,
                           __bf16* __restrict__ decf_out) {
  int b = blockIdx.x * blockDim.x + threadIdx.x;
  if (b >= B_) return;
  float vals[C_];
  float sum = 0.f, sq = 0.f;
#pragma unroll
  for (int j = 0; j < C_; ++j) {
    float d = delta[(size_t)b * FCNP + j] + fcb[j];
    vals[j] = d;
    sum += d;
    sq  += d * d;
  }
  float m   = sum * (1.0f / C_);
  float var = sq * (1.0f / C_) - m * m;
  float rs  = rsqrtf(var + 1e-5f);
#pragma unroll
  for (int j = 0; j < C_; ++j) {
    float nv = (vals[j] - m) * rs * alpha[j] + beta[j] + dec[(size_t)b * C_ + j];
    out[((size_t)b * T_ + s) * C_ + j] = nv;
    dec[(size_t)b * C_ + j] = nv;
    size_t o = (((size_t)(b >> 4) * (KXP / 32) + (j >> 5)) << 9) + frag_pos(b & 15, j & 31);
    decf_out[o] = (__bf16)nv;
  }
}

// --------------------------------- host ------------------------------------

extern "C" void kernel_launch(void* const* d_in, const int* in_sizes, int n_in,
                              void* d_out, int out_size, void* d_ws, size_t ws_size,
                              hipStream_t stream) {
  const float* x     = (const float*)d_in[0];
  const float* W_ih0 = (const float*)d_in[1];
  const float* W_hh0 = (const float*)d_in[2];
  const float* b0    = (const float*)d_in[3];
  const float* W_ih1 = (const float*)d_in[4];
  const float* W_hh1 = (const float*)d_in[5];
  const float* b1    = (const float*)d_in[6];
  const float* fc_w  = (const float*)d_in[7];
  const float* fc_b  = (const float*)d_in[8];
  const float* alpha = (const float*)d_in[9];
  const float* beta  = (const float*)d_in[10];
  float* out = (float*)d_out;

  size_t off = 0;
  auto carve = [&](size_t bytes) -> char* {
    char* r = (char*)d_ws + off;
    off += (bytes + 255) & ~(size_t)255;
    return r;
  };
  const size_t FR = 512 * sizeof(__bf16);
  __bf16* Wih0s = (__bf16*)carve((N4 / 16) * (KXP / 32) * FR);
  __bf16* Whh0s = (__bf16*)carve((N4 / 16) * (H_ / 32) * FR);
  __bf16* Wih1s = (__bf16*)carve((N4 / 16) * (H_ / 32) * FR);
  __bf16* Whh1s = (__bf16*)carve((N4 / 16) * (H_ / 32) * FR);
  __bf16* fcws  = (__bf16*)carve((FCNP / 16) * (H_ / 32) * FR);
  __bf16* xs    = (__bf16*)carve((size_t)T_ * (B_ / 16) * (KXP / 32) * FR);
  float*  delta = (float*)carve((size_t)B_ * FCNP * 4);
  float*  dec   = (float*)carve((size_t)B_ * C_ * 4);
  size_t zbeg = off;
  float*  c0 = (float*)carve((size_t)B_ * H_ * 4);
  float*  c1 = (float*)carve((size_t)B_ * H_ * 4);
  __bf16* h0f[2] = { (__bf16*)carve((B_ / 16) * (H_ / 32) * FR),
                     (__bf16*)carve((B_ / 16) * (H_ / 32) * FR) };
  __bf16* h1f[2] = { (__bf16*)carve((B_ / 16) * (H_ / 32) * FR),
                     (__bf16*)carve((B_ / 16) * (H_ / 32) * FR) };
  __bf16* decf[2] = { (__bf16*)carve((B_ / 16) * (KXP / 32) * FR),
                      (__bf16*)carve((B_ / 16) * (KXP / 32) * FR) };
  size_t zend = off;
  (void)ws_size; (void)in_sizes; (void)n_in; (void)out_size;

  // prologue
  zero_u32<<<2048, 256, 0, stream>>>((unsigned*)((char*)d_ws + zbeg),
                                     (long)((zend - zbeg) / 4));
  scatter_w<<<(N4 * KXP + 255) / 256, 256, 0, stream>>>(W_ih0, N4, N4, C_, KXP, Wih0s);
  scatter_w<<<(N4 * H_ + 255) / 256, 256, 0, stream>>>(W_hh0, N4, N4, H_, H_, Whh0s);
  scatter_w<<<(N4 * H_ + 255) / 256, 256, 0, stream>>>(W_ih1, N4, N4, H_, H_, Wih1s);
  scatter_w<<<(N4 * H_ + 255) / 256, 256, 0, stream>>>(W_hh1, N4, N4, H_, H_, Whh1s);
  scatter_w<<<(FCNP * H_ + 255) / 256, 256, 0, stream>>>(fc_w, C_, FCNP, H_, H_, fcws);
  scatter_x<<<(int)(((long)B_ * T_ * KXP + 255) / 256), 256, 0, stream>>>(x, xs);
  init_dec<<<(B_ * C_ + 255) / 256, 256, 0, stream>>>(x, dec);

  const int LGRID = (B_ / 128) * (H_ / 16);                  // 512 blocks
  const size_t xstep = (size_t)(B_ / 16) * (KXP / 32) * 512; // elems per t
  int p = 0;

  // encoder: t = 0 .. T-2
  for (int t = 0; t < T_ - 1; ++t) {
    lstm_layer<KXP / 32><<<LGRID, 128, 0, stream>>>(xs + (size_t)t * xstep,
                                                    h0f[p], Wih0s, Whh0s, b0, c0, h0f[1 - p]);
    lstm_layer<H_ / 32><<<LGRID, 128, 0, stream>>>(h0f[1 - p],
                                                   h1f[p], Wih1s, Whh1s, b1, c1, h1f[1 - p]);
    p ^= 1;
  }

  // decoder: s = 0 .. T-1
  for (int s = 0; s < T_; ++s) {
    const __bf16* Ax = (s == 0) ? (xs + (size_t)(T_ - 1) * xstep) : decf[s & 1];
    lstm_layer<KXP / 32><<<LGRID, 128, 0, stream>>>(Ax,
                                                    h0f[p], Wih0s, Whh0s, b0, c0, h0f[1 - p]);
    lstm_layer<H_ / 32><<<LGRID, 128, 0, stream>>>(h0f[1 - p],
                                                   h1f[p], Wih1s, Whh1s, b1, c1, h1f[1 - p]);
    fc_gemm<<<dim3(B_ / 64, FCNP / 16), 128, 0, stream>>>(h1f[1 - p], fcws, delta);
    dec_finish<<<(B_ + 255) / 256, 256, 0, stream>>>(delta, fc_b, alpha, beta,
                                                     dec, out, s, decf[(s + 1) & 1]);
    p ^= 1;
  }
}
